// ResNet_bngn_dynamic_37503654429270
// MI455X (gfx1250) — matrix operand order repaired and checked by reading, not compiled, not run on hardware
//
#include <hip/hip_runtime.h>
#include <hip/hip_bf16.h>
#include <math.h>

// ---------------------------------------------------------------------------
// CDNA5 (gfx1250) implementation of the MoE ResNet forward pass.
// All convolutions run as implicit GEMM on v_wmma_f32_16x16x32_bf16, with
// double-buffered LDS, async global->LDS DMA for the weight tiles (ASYNCcnt),
// and register-staged im2col gathers overlapping the WMMA issue.
// Routing: forward mask is exactly one-hot (Sterbenz), so each MoE conv launch
// is gated per-sample (grid.z = batch) and only the argmax expert runs.
// ---------------------------------------------------------------------------

typedef __bf16 bf16_t;
typedef bf16_t v16bf __attribute__((ext_vector_type(16)));
typedef float  v8f   __attribute__((ext_vector_type(8)));

#define EPS 1e-5f
#define NB 6           // number of routed (MoE) blocks
#define BATCH 32

#if defined(__has_builtin)
#if __has_builtin(__builtin_amdgcn_global_load_async_to_lds_b64) && \
    __has_builtin(__builtin_amdgcn_s_wait_asynccnt)
#define USE_ASYNC_LDS 1
#endif
#endif
#ifndef USE_ASYNC_LDS
#define USE_ASYNC_LDS 0
#endif

#if USE_ASYNC_LDS
// builtin signature (from hipcc diagnostic): (v2i AS1*, v2i AS3*, Ii, Ii)
typedef int v2i_t __attribute__((ext_vector_type(2)));
typedef __attribute__((address_space(1))) v2i_t as1_v2i;
typedef __attribute__((address_space(3))) v2i_t as3_v2i;
#endif

union PairPack { bf16_t h[2]; unsigned u; };
union FragPack { v16bf v; unsigned u[8]; };

// ---------------------------------------------------------------------------
// One-shot weight pre-conversion: f32 [Cout][KK] -> bf16 [Cout][KKp],
// KKp = round_up(KK,32), zero-filled pad. Removes all per-tile conversion
// work and K-bounds checks from the conv A-path, and makes the A tile a
// pure (async-DMA-able) copy.
// ---------------------------------------------------------------------------
__global__ __launch_bounds__(256)
void cvt_weight_kernel(const float* __restrict__ Wf, bf16_t* __restrict__ Wb,
                       int KK, int KKp, int total)
{
    int idx = blockIdx.x * 256 + threadIdx.x;
    if (idx >= total) return;
    int c = idx / KKp, k = idx - c * KKp;
    float v = (k < KK) ? Wf[(size_t)c * KK + k] : 0.f;
    Wb[idx] = (bf16_t)v;
}

// ---------------------------------------------------------------------------
// Implicit-GEMM conv:  Y[b][co][oh][ow] = sum_{ci,kh,kw} ...
// GEMM view: M=Cout, N=OH*OW (per image), K=Cin*KH*KW (padded to KKp).
// Workgroup = 128 threads = 4 waves, 64(M) x 64(N) tile; each wave owns a
// 32x32 sub-tile (2 A frags x 2 B frags -> 4 WMMAs/K-step, full reuse).
// Pipeline (1 barrier per K-step, LDS double-buffered):
//   s_wait_asynccnt 0 ; barrier
//   issue async A-DMA(next) + B gather loads(next) into regs
//   ds-read fragments + 4x WMMA (current)     <- overlaps global latency
//   cvt + ds-store B(next)
// LDS rows padded to 36 elems (72B): 8B-aligned for b64 async stores and
// bank-conflict-free for the 4B pair-wise fragment reads.
// route!=null => whole workgroup exits unless sample b is routed to `expert`.
// ---------------------------------------------------------------------------
template<int KH, int KW>
__global__ __launch_bounds__(128)
void conv_wmma_kernel(const float* __restrict__ X,
                      const bf16_t* __restrict__ Wb,   // [Cout][KKp] bf16
                      float* __restrict__ Y,
                      const float* __restrict__ bn_g, const float* __restrict__ bn_b,
                      const float* __restrict__ bn_m, const float* __restrict__ bn_v,
                      const int* __restrict__ route, int moe_idx, int expert,
                      int Cin, int H, int Wd,
                      int Cout, int OH, int OW,
                      int stride, int pad, int KKp,
                      int relu_in, int relu_out)
{
    constexpr int KHW = KH * KW;
    const int bz = blockIdx.z;                  // batch sample
    if (route) {                                // wave-uniform expert gate
        if (route[bz * NB + moe_idx] != expert) return;
    }
    const int tid  = threadIdx.x;
    const int lane = tid & 31;
    const int wave = tid >> 5;
    const int wm = wave >> 1;                   // 0..1 (32-row M sub-tile)
    const int wn = wave & 1;                    // 0..1 (32-col N sub-tile)
    const int half = lane >> 4;                 // lane 0-15 vs 16-31
    const int l16  = lane & 15;

    const int mbase = blockIdx.y * 64;
    const int nbase = blockIdx.x * 64;
    const int OHW = OH * OW;
    const int KK = Cin * KHW;
    const int ksteps = KKp >> 5;

    __shared__ bf16_t Asm[2][64][36];   // [buf][m_local][k]  weights
    __shared__ bf16_t Bsm[2][64][36];   // [buf][n_local][k]  im2col patches

    v8f acc[2][2] = {};                 // [m frag][n frag]

    uint2 areg[4];
    float bx0[8], bx1[8];

    const float* xb = X + (size_t)bz * Cin * H * Wd;

    auto stageA = [&](int ks, int buf) {
        #pragma unroll
        for (int c = 0; c < 4; ++c) {
            int chunk = tid * 4 + c;            // A tile = 512 8B chunks
            int mi = chunk >> 3, j = chunk & 7;
            const bf16_t* src = Wb + (size_t)(mbase + mi) * KKp + (ks << 5) + j * 4;
#if USE_ASYNC_LDS
            __builtin_amdgcn_global_load_async_to_lds_b64(
                (as1_v2i*)src, (as3_v2i*)&Asm[buf][mi][j * 4], 0, 0);
#else
            areg[c] = *(const uint2*)src;
#endif
        }
        (void)buf;
    };
    auto storeA = [&](int buf) {
#if !USE_ASYNC_LDS
        #pragma unroll
        for (int c = 0; c < 4; ++c) {
            int chunk = tid * 4 + c;
            int mi = chunk >> 3, j = chunk & 7;
            *(uint2*)&Asm[buf][mi][j * 4] = areg[c];
        }
#else
        (void)buf;
#endif
    };
    auto gatherB = [&](int ks) {
        #pragma unroll
        for (int t = 0; t < 8; ++t) {
            int pidx = t * 128 + tid;           // B tile = 1024 pairs
            int ni = pidx >> 4, kp = (pidx & 15) << 1;
            int ng = nbase + ni;
            float x0 = 0.f, x1 = 0.f;
            if (ng < OHW) {
                int oh = ng / OW, ow = ng - oh * OW;
                int ihb = oh * stride - pad, iwb = ow * stride - pad;
                #pragma unroll
                for (int e = 0; e < 2; ++e) {
                    int kg = (ks << 5) + kp + e;
                    if (kg < KK) {
                        int ci  = kg / KHW;          // compile-time divisor
                        int rem = kg - ci * KHW;
                        int kh = rem / KW, kw = rem - kh * KW;
                        int ih = ihb + kh, iw = iwb + kw;
                        if ((unsigned)ih < (unsigned)H && (unsigned)iw < (unsigned)Wd) {
                            float xv = xb[((size_t)ci * H + ih) * Wd + iw];
                            if (relu_in) xv = fmaxf(xv, 0.f);
                            if (e) x1 = xv; else x0 = xv;
                        }
                    }
                }
            }
            bx0[t] = x0; bx1[t] = x1;
        }
    };
    auto storeB = [&](int buf) {
        #pragma unroll
        for (int t = 0; t < 8; ++t) {
            int pidx = t * 128 + tid;
            int ni = pidx >> 4, kp = (pidx & 15) << 1;
            PairPack p; p.h[0] = (bf16_t)bx0[t]; p.h[1] = (bf16_t)bx1[t];
            *(unsigned*)&Bsm[buf][ni][kp] = p.u;
        }
    };

    // ---- prologue: stage step 0 into buffer 0 ------------------------------
    stageA(0, 0);
    gatherB(0);
    storeA(0);
    storeB(0);

    for (int ks = 0; ks < ksteps; ++ks) {
        const int buf = ks & 1;
#if USE_ASYNC_LDS
        __builtin_amdgcn_s_wait_asynccnt(0);    // my A-DMA into `buf` done
#endif
        __syncthreads();                        // everyone's stores visible

        const bool has_next = (ks + 1 < ksteps);
        if (has_next) {                         // issue next-step globals early
            stageA(ks + 1, buf ^ 1);
            gatherB(ks + 1);
        }

        // ---- fragments per documented 16-bit WMMA layouts ------------------
        // A (16x32): VGPR v<4: K = 2v+e + 8*half ; v>=4: K = 16+2(v-4)+e + 8*half
        // B (32x16): same K order as A: v<4: K = 2v+e + 8*half ; v>=4: K = 16+2(v-4)+e + 8*half
        FragPack af[2], bfr[2];
        #pragma unroll
        for (int mi = 0; mi < 2; ++mi) {
            int mrow = wm * 32 + mi * 16 + l16;
            #pragma unroll
            for (int v = 0; v < 8; ++v) {
                int ka = ((v >= 4) ? 16 : 0) + half * 8 + (v & 3) * 2;
                af[mi].u[v] = *(const unsigned*)&Asm[buf][mrow][ka];
            }
        }
        #pragma unroll
        for (int nj = 0; nj < 2; ++nj) {
            int nrow = wn * 32 + nj * 16 + l16;
            #pragma unroll
            for (int v = 0; v < 8; ++v) {
                int kb = ((v >= 4) ? 16 : 0) + half * 8 + (v & 3) * 2;
                bfr[nj].u[v] = *(const unsigned*)&Bsm[buf][nrow][kb];
            }
        }
        #pragma unroll
        for (int mi = 0; mi < 2; ++mi)
            #pragma unroll
            for (int nj = 0; nj < 2; ++nj)
                acc[mi][nj] = __builtin_amdgcn_wmma_f32_16x16x32_bf16(
                    false, af[mi].v, false, bfr[nj].v,
                    (short)0, acc[mi][nj], false, false);

        if (has_next) {                         // complete next-step staging
            storeA(buf ^ 1);
            storeB(buf ^ 1);
        }
    }

    // ---- epilogue: fused BN + ReLU, NCHW scatter ---------------------------
    float sc[2][8], bs[2][8];
    #pragma unroll
    for (int mi = 0; mi < 2; ++mi)
        #pragma unroll
        for (int r = 0; r < 8; ++r) {
            int c = mbase + wm * 32 + mi * 16 + r + half * 8;
            float s = 1.f, o = 0.f;
            if (bn_g) {
                float inv = bn_g[c] * rsqrtf(bn_v[c] + EPS);
                s = inv; o = bn_b[c] - bn_m[c] * inv;
            }
            sc[mi][r] = s; bs[mi][r] = o;
        }
    #pragma unroll
    for (int nj = 0; nj < 2; ++nj) {
        int ng = nbase + wn * 32 + nj * 16 + l16;
        if (ng < OHW) {
            int oh = ng / OW, ow = ng - oh * OW;
            #pragma unroll
            for (int mi = 0; mi < 2; ++mi)
                #pragma unroll
                for (int r = 0; r < 8; ++r) {
                    int c = mbase + wm * 32 + mi * 16 + r + half * 8;
                    float val = acc[mi][nj][r] * sc[mi][r] + bs[mi][r];
                    if (relu_out) val = fmaxf(val, 0.f);
                    Y[(((size_t)bz * Cout + c) * OH + oh) * OW + ow] = val;
                }
        }
    }
}

// ---------------------------------------------------------------------------
// Maxpool 3x3 stride 2 pad 1 (memory-bound, trivial)
// ---------------------------------------------------------------------------
__global__ __launch_bounds__(256)
void maxpool3s2_kernel(const float* __restrict__ X, float* __restrict__ Y,
                       int C, int H, int W, int OH, int OW, int total)
{
    int idx = blockIdx.x * 256 + threadIdx.x;
    if (idx >= total) return;
    int ow = idx % OW; int t = idx / OW;
    int oh = t % OH;   t /= OH;
    int c  = t % C;    int b = t / C;
    float m = -INFINITY;
    #pragma unroll
    for (int dy = 0; dy < 3; ++dy) {
        int ih = oh * 2 - 1 + dy;
        if ((unsigned)ih >= (unsigned)H) continue;
        #pragma unroll
        for (int dx = 0; dx < 3; ++dx) {
            int iw = ow * 2 - 1 + dx;
            if ((unsigned)iw >= (unsigned)W) continue;
            m = fmaxf(m, X[(((size_t)b * C + c) * H + ih) * W + iw]);
        }
    }
    Y[idx] = m;
}

// ---------------------------------------------------------------------------
// Router: avgpool7 (56x56 -> 8x8, 64ch) -> FC 4096->64 + ReLU -> FC 64->12
// -> per (b, block) argmax over E=2 (argmax of softmax == argmax of logits;
//    ties pick index 0, matching jnp.argmax).  One workgroup per sample.
// ---------------------------------------------------------------------------
__global__ __launch_bounds__(256)
void router_kernel(const float* __restrict__ H0,
                   const float* __restrict__ W1, const float* __restrict__ B1,
                   const float* __restrict__ W2, const float* __restrict__ B2,
                   int* __restrict__ route)
{
    __shared__ float pool[4096];
    __shared__ float p1[64];
    __shared__ float logits[12];
    const int b = blockIdx.x, tid = threadIdx.x;

    for (int i = tid; i < 4096; i += 256) {       // i = c*64 + oh*8 + ow
        int c = i >> 6, oh = (i >> 3) & 7, ow = i & 7;
        float s = 0.f;
        for (int dy = 0; dy < 7; ++dy)
            for (int dx = 0; dx < 7; ++dx)
                s += H0[(((size_t)b * 64 + c) * 56 + oh * 7 + dy) * 56 + ow * 7 + dx];
        pool[i] = s * (1.f / 49.f);
    }
    __syncthreads();
    if (tid < 64) {
        float s = B1[tid];
        for (int k = 0; k < 4096; ++k) s += pool[k] * W1[(size_t)tid * 4096 + k];
        p1[tid] = fmaxf(s, 0.f);
    }
    __syncthreads();
    if (tid < 12) {
        float s = B2[tid];
        for (int k = 0; k < 64; ++k) s += p1[k] * W2[tid * 64 + k];
        logits[tid] = s;
    }
    __syncthreads();
    if (tid < NB) route[b * NB + tid] = (logits[2 * tid + 1] > logits[2 * tid]) ? 1 : 0;
}

// ---------------------------------------------------------------------------
// GroupNorm (groups=4) pass 1: per (b, group) mean and rsqrt(var+eps)
// ---------------------------------------------------------------------------
__global__ __launch_bounds__(256)
void gn_stats_kernel(const float* __restrict__ X, float* __restrict__ stats,
                     int C, int HW)
{
    const int tid = threadIdx.x;
    const int b = blockIdx.x >> 2, g = blockIdx.x & 3;
    const int Cg = C >> 2;
    const size_t base = ((size_t)b * C + (size_t)g * Cg) * HW;
    const int n = Cg * HW;
    float s = 0.f, s2 = 0.f;
    for (int i = tid; i < n; i += 256) {
        float v = X[base + i];
        s += v; s2 += v * v;
    }
    __shared__ float sh[256], sh2[256];
    sh[tid] = s; sh2[tid] = s2;
    __syncthreads();
    for (int o = 128; o > 0; o >>= 1) {
        if (tid < o) { sh[tid] += sh[tid + o]; sh2[tid] += sh2[tid + o]; }
        __syncthreads();
    }
    if (tid == 0) {
        float mu  = sh[0] / n;
        float var = sh2[0] / n - mu * mu;
        stats[blockIdx.x * 2]     = mu;
        stats[blockIdx.x * 2 + 1] = rsqrtf(var + EPS);
    }
}

// ---------------------------------------------------------------------------
// GN pass 2 fused with residual add + ReLU:
//   Y = relu( (X - mu)*inv * g[c] + b[c] + identity )
// ---------------------------------------------------------------------------
__global__ __launch_bounds__(256)
void gn_apply_kernel(const float* __restrict__ X, const float* __restrict__ stats,
                     const float* __restrict__ gw, const float* __restrict__ gb,
                     const float* __restrict__ ident, float* __restrict__ Y,
                     int C, int HW, int total)
{
    int idx = blockIdx.x * 256 + threadIdx.x;
    if (idx >= total) return;
    int bc = idx / HW;
    int c = bc % C, b = bc / C;
    int g = c / (C >> 2);
    float mu  = stats[(b * 4 + g) * 2];
    float inv = stats[(b * 4 + g) * 2 + 1];
    float v = (X[idx] - mu) * inv * gw[c] + gb[c] + ident[idx];
    Y[idx] = fmaxf(v, 0.f);
}

// ---------------------------------------------------------------------------
// Head: spatial mean pool, then FC 512->1000
// ---------------------------------------------------------------------------
__global__ __launch_bounds__(256)
void avgpool_hw_kernel(const float* __restrict__ X, float* __restrict__ P,
                       int HW, int total)
{
    int idx = blockIdx.x * 256 + threadIdx.x;
    if (idx >= total) return;
    float s = 0.f;
    for (int i = 0; i < HW; ++i) s += X[(size_t)idx * HW + i];
    P[idx] = s / (float)HW;
}

__global__ __launch_bounds__(256)
void fc_kernel(const float* __restrict__ P, const float* __restrict__ Wt,
               const float* __restrict__ Bb, float* __restrict__ Y,
               int Cin, int Cout, int total)
{
    int idx = blockIdx.x * 256 + threadIdx.x;
    if (idx >= total) return;
    int c = idx % Cout, b = idx / Cout;
    float s = Bb[c];
    for (int k = 0; k < Cin; ++k) s += P[(size_t)b * Cin + k] * Wt[(size_t)c * Cin + k];
    Y[idx] = s;
}

// ---------------------------------------------------------------------------
// Host side
// ---------------------------------------------------------------------------
static inline void conv(hipStream_t st, const float* X, const float* W, float* Y,
                        bf16_t* wbuf,
                        const float* g, const float* b, const float* m, const float* v,
                        const int* route, int moe, int exp_,
                        int Cin, int H, int Wd, int Cout, int OH, int OW,
                        int KH, int KW, int stride, int pad, int relu_in, int relu_out)
{
    int KK  = Cin * KH * KW;
    int KKp = (KK + 31) & ~31;
    int tot = Cout * KKp;
    cvt_weight_kernel<<<(tot + 255) / 256, 256, 0, st>>>(W, wbuf, KK, KKp, tot);

    dim3 grid((OH * OW + 63) / 64, (Cout + 63) / 64, BATCH);
    if (KH == 3)
        conv_wmma_kernel<3,3><<<grid, 128, 0, st>>>(X, wbuf, Y, g, b, m, v, route, moe, exp_,
            Cin, H, Wd, Cout, OH, OW, stride, pad, KKp, relu_in, relu_out);
    else if (KH == 1)
        conv_wmma_kernel<1,1><<<grid, 128, 0, st>>>(X, wbuf, Y, g, b, m, v, route, moe, exp_,
            Cin, H, Wd, Cout, OH, OW, stride, pad, KKp, relu_in, relu_out);
    else
        conv_wmma_kernel<7,7><<<grid, 128, 0, st>>>(X, wbuf, Y, g, b, m, v, route, moe, exp_,
            Cin, H, Wd, Cout, OH, OW, stride, pad, KKp, relu_in, relu_out);
}

extern "C" void kernel_launch(void* const* d_in, const int* in_sizes, int n_in,
                              void* d_out, int out_size, void* d_ws, size_t ws_size,
                              hipStream_t stream)
{
    (void)in_sizes; (void)n_in; (void)out_size; (void)ws_size;
    auto F = [&](int i) { return (const float*)d_in[i]; };

    // ---- workspace carve-out ----------------------------------------------
    char* w = (char*)d_ws;
    size_t off = 0;
    auto alloc = [&](size_t elems) { float* p = (float*)(w + off); off += elems * 4; return p; };
    float* stem   = alloc((size_t)BATCH * 64 * 112 * 112);  // stem conv output
    float* hA     = alloc((size_t)BATCH * 64 * 56 * 56);    // ping
    float* hB     = alloc((size_t)BATCH * 64 * 56 * 56);    // pong
    float* t1     = alloc((size_t)BATCH * 64 * 56 * 56);    // expert conv1 out
    float* bo     = alloc((size_t)BATCH * 64 * 56 * 56);    // block out (pre-GN)
    float* idb    = alloc((size_t)BATCH * 128 * 28 * 28);   // downsample identity
    float* stats  = alloc(BATCH * 4 * 2);                   // GN mean/inv
    float* pooled = alloc(BATCH * 512);                     // head pool
    int*   route  = (int*)(w + off); off += BATCH * NB * 4;
    off = (off + 255) & ~(size_t)255;                       // align bf16 panel
    bf16_t* wbuf  = (bf16_t*)(w + off);                     // largest: 512x4608
    off += (size_t)512 * 4608 * 2;

    // ---- stem: conv7x7 s2 p3 + BN + ReLU (WMMA implicit GEMM) -------------
    conv(stream, F(0), F(1), stem, wbuf, F(2), F(3), F(4), F(5), nullptr, 0, 0,
         3, 224, 224, 64, 112, 112, 7, 7, 2, 3, /*relu_in*/0, /*relu_out*/1);

    // ---- maxpool 3x3 s2 p1 -> hA ------------------------------------------
    {
        int total = BATCH * 64 * 56 * 56;
        maxpool3s2_kernel<<<(total + 255) / 256, 256, 0, stream>>>(
            stem, hA, 64, 112, 112, 56, 56, total);
    }

    // ---- router (per-sample argmax expert per MoE block) ------------------
    router_kernel<<<BATCH, 256, 0, stream>>>(hA, F(121), F(122), F(123), F(124), route);

    // ---- residual blocks ---------------------------------------------------
    // Flattened d_in index bases per the dict insertion order of setup_inputs.
    struct Blk { int base, nexp, has_ds, cin, cout, hin, hout, stride, moe; };
    const Blk blks[8] = {
        {  6, 1, 0,  64,  64, 56, 56, 1, -1 },   // L0B0
        { 14, 1, 0,  64,  64, 56, 56, 1, -1 },   // L0B1
        { 22, 2, 1,  64, 128, 56, 28, 2,  0 },   // L1B0 (ds)
        { 41, 2, 0, 128, 128, 28, 28, 1,  1 },   // L1B1
        { 55, 2, 1, 128, 256, 28, 14, 2,  2 },   // L2B0 (ds)
        { 74, 2, 0, 256, 256, 14, 14, 1,  3 },   // L2B1
        { 88, 2, 1, 256, 512, 14,  7, 2,  4 },   // L3B0 (ds)
        {107, 2, 0, 512, 512,  7,  7, 1,  5 },   // L3B1
    };

    float* cur = hA;
    float* nxt = hB;
    for (int bi = 0; bi < 8; ++bi) {
        const Blk& k = blks[bi];
        const int* rt = (k.moe >= 0) ? route : nullptr;

        for (int e = 0; e < k.nexp; ++e) {
            int wb = k.base + e * 6;     // w1, bn1{g,b,m,v}, w2
            // conv1: relu(x) -> 3x3 conv -> BN -> ReLU  (all fused)
            conv(stream, cur, F(wb), t1, wbuf, F(wb + 1), F(wb + 2), F(wb + 3), F(wb + 4),
                 rt, k.moe, e,
                 k.cin, k.hin, k.hin, k.cout, k.hout, k.hout,
                 3, 3, k.stride, 1, /*relu_in*/1, /*relu_out*/1);
            // conv2: raw output (GN follows)
            conv(stream, t1, F(wb + 5), bo, wbuf, nullptr, nullptr, nullptr, nullptr,
                 rt, k.moe, e,
                 k.cout, k.hout, k.hout, k.cout, k.hout, k.hout,
                 3, 3, 1, 1, /*relu_in*/0, /*relu_out*/0);
        }

        const float* ident = cur;
        if (k.has_ds) {
            int db = k.base + k.nexp * 6 + 2;   // ds_w, ds_bn{g,b,m,v}
            conv(stream, cur, F(db), idb, wbuf, F(db + 1), F(db + 2), F(db + 3), F(db + 4),
                 nullptr, 0, 0,
                 k.cin, k.hin, k.hin, k.cout, k.hout, k.hout,
                 1, 1, k.stride, 0, /*relu_in*/1, /*relu_out*/0);
            ident = idb;
        }

        int HW = k.hout * k.hout;
        gn_stats_kernel<<<BATCH * 4, 256, 0, stream>>>(bo, stats, k.cout, HW);
        int total = BATCH * k.cout * HW;
        gn_apply_kernel<<<(total + 255) / 256, 256, 0, stream>>>(
            bo, stats, F(k.base + k.nexp * 6), F(k.base + k.nexp * 6 + 1),
            ident, nxt, k.cout, HW, total);

        float* tmp = cur; cur = nxt; nxt = tmp;
    }

    // ---- head: mean over 7x7, FC 512->1000 --------------------------------
    {
        int tp = BATCH * 512;
        avgpool_hw_kernel<<<(tp + 255) / 256, 256, 0, stream>>>(cur, pooled, 49, tp);
        int tf = BATCH * 1000;
        fc_kernel<<<(tf + 255) / 256, 256, 0, stream>>>(
            pooled, F(125), F(126), (float*)d_out, 512, 1000, tf);
    }
}
